// BiLSTM_CRF_369367187863
// MI455X (gfx1250) — compile-verified, matrix operand order checked by reading, not run on hardware
//
#include <hip/hip_runtime.h>
#include <hip/hip_bf16.h>
#include <math.h>

typedef __attribute__((ext_vector_type(16))) _Float16 v16h;
typedef __attribute__((ext_vector_type(8)))  _Float16 v8h;
typedef __attribute__((ext_vector_type(8)))  float    v8f;

#define L_SEQ 4096
#define EDIM  256
#define H2D   256
#define G4    1024   // 4*H2
#define TAGS  20
#define NEGV  -10000.0f

union Frag16 { v16h v; v8h h8[2]; };

// ---------------------------------------------------------------------------
// Phase 1a: embedding gather (f32 -> f16) + W_ih f16 conversion (both dirs)
// ---------------------------------------------------------------------------
__global__ void prep_kernel(const int* __restrict__ sentence,
                            const float* __restrict__ table,
                            const float* __restrict__ Wf,
                            const float* __restrict__ Wb,
                            _Float16* __restrict__ emb_h,
                            _Float16* __restrict__ Wf_h,
                            _Float16* __restrict__ Wb_h) {
  const int NE = L_SEQ * EDIM;     // 1,048,576
  const int NW = G4 * EDIM;        // 262,144
  int idx = blockIdx.x * blockDim.x + threadIdx.x;
  if (idx < NE) {
    int t = idx >> 8, e = idx & 255;
    emb_h[idx] = (_Float16)table[(long long)sentence[t] * EDIM + e];
  } else if (idx < NE + NW) {
    int i = idx - NE;
    Wf_h[i] = (_Float16)Wf[i];
  } else if (idx < NE + 2 * NW) {
    int i = idx - NE - NW;
    Wb_h[i] = (_Float16)Wb[i];
  }
}

// ---------------------------------------------------------------------------
// Phase 1b: input projection xp = emb @ W_ih^T + (b_ih + b_hh), both dirs.
// One wave per 16x16 output tile, K looped 256 in steps of 32 via
// v_wmma_f32_16x16x32_f16. dir=1 uses time-reversed embedding rows.
// ---------------------------------------------------------------------------
__global__ void __launch_bounds__(256)
inproj_kernel(const _Float16* __restrict__ emb_h,
              const _Float16* __restrict__ Wf_h,
              const _Float16* __restrict__ Wb_h,
              const float* __restrict__ b_ih_f, const float* __restrict__ b_hh_f,
              const float* __restrict__ b_ih_b, const float* __restrict__ b_hh_b,
              float* __restrict__ xp_f, float* __restrict__ xp_b) {
  const int wave = (blockIdx.x * blockDim.x + threadIdx.x) >> 5;
  const int lane = threadIdx.x & 31;
  // tiles: dir(2) x tm(256) x tn(64)  -> 32768 waves total
  const int dir = wave >> 14;
  const int rem = wave & 16383;
  const int tm  = rem >> 6;
  const int tn  = rem & 63;

  const _Float16* __restrict__ W = dir ? Wb_h : Wf_h;
  const int l16 = lane & 15;
  const int hi8 = (lane >> 4) << 3;                 // 0 or 8
  const int row  = tm * 16 + l16;                   // output time row (A)
  const int arow = dir ? (L_SEQ - 1 - row) : row;   // reversed emb for bwd dir
  const int ncol = tn * 16 + l16;                   // output gate column (B/D)

  const _Float16* aBase = emb_h + (long long)arow * EDIM;
  const _Float16* bBase = W     + (long long)ncol * EDIM; // row-major W == col-major B

  v8f acc = {};
#pragma unroll
  for (int k0 = 0; k0 < EDIM; k0 += 32) {
    Frag16 a, b;
    a.h8[0] = *(const v8h*)(aBase + k0 + hi8);
    a.h8[1] = *(const v8h*)(aBase + k0 + 16 + hi8);
    b.h8[0] = *(const v8h*)(bBase + k0 + hi8);
    b.h8[1] = *(const v8h*)(bBase + k0 + 16 + hi8);
    acc = __builtin_amdgcn_wmma_f32_16x16x32_f16(false, a.v, false, b.v,
                                                 (short)0, acc, false, false);
  }

  float* __restrict__ xp = dir ? xp_b : xp_f;
  const float bias = dir ? (b_ih_b[ncol] + b_hh_b[ncol])
                         : (b_ih_f[ncol] + b_hh_f[ncol]);
  const int mbase = tm * 16 + ((lane >> 4) << 3);   // D: VGPR r -> row mbase+r
#pragma unroll
  for (int r = 0; r < 8; ++r)
    xp[(long long)(mbase + r) * G4 + ncol] = acc[r] + bias;
}

// ---------------------------------------------------------------------------
// Phase 2: sequential LSTM scan. One 1024-thread block per direction.
// W_hh (1MB/dir) stays L2-resident; h/c/z live in LDS (6KB).
// ---------------------------------------------------------------------------
__device__ __forceinline__ float sigm_(float x) { return 1.0f / (1.0f + __expf(-x)); }

__global__ void __launch_bounds__(1024)
lstm_scan_kernel(const float* __restrict__ xp_f, const float* __restrict__ xp_b,
                 const float* __restrict__ W_hh_f, const float* __restrict__ W_hh_b,
                 const float* __restrict__ h0, const float* __restrict__ c0,
                 float* __restrict__ hs_f, float* __restrict__ hs_b) {
  __shared__ float h_s[H2D];
  __shared__ float c_s[H2D];
  __shared__ float z_s[G4];
  const int dir = blockIdx.x;
  const float* __restrict__ xp = dir ? xp_b : xp_f;
  const float* __restrict__ W  = dir ? W_hh_b : W_hh_f;
  float* __restrict__ hs       = dir ? hs_b : hs_f;
  const int j = threadIdx.x;
  if (j < H2D) { h_s[j] = h0[dir * H2D + j]; c_s[j] = c0[dir * H2D + j]; }
  __syncthreads();
  const float* __restrict__ wrow = W + (long long)j * H2D;
  for (int t = 0; t < L_SEQ; ++t) {
    __builtin_prefetch(&xp[(long long)(t + 1) * G4 + j], 0, 1); // next step xp
    float acc = xp[(long long)t * G4 + j];
#pragma unroll 8
    for (int k = 0; k < H2D; ++k) acc += h_s[k] * wrow[k];
    z_s[j] = acc;
    __syncthreads();          // all dots done before h/c update
    if (j < H2D) {
      float ig = sigm_(z_s[j]);
      float fg = sigm_(z_s[H2D + j]);
      float gg = tanhf(z_s[2 * H2D + j]);
      float og = sigm_(z_s[3 * H2D + j]);
      float c  = fg * c_s[j] + ig * gg;
      c_s[j] = c;
      float h = og * tanhf(c);
      h_s[j] = h;
      hs[(long long)t * H2D + j] = h;
    }
    __syncthreads();          // h_s stable before next step's dots
  }
}

// ---------------------------------------------------------------------------
// Phase 3: feats = [hf, hb] @ W_out^T + b_out.  hb[t] = hs_b[L-1-t].
// ---------------------------------------------------------------------------
__global__ void outproj_kernel(const float* __restrict__ hs_f,
                               const float* __restrict__ hs_b,
                               const float* __restrict__ W_out,
                               const float* __restrict__ b_out,
                               float* __restrict__ feats) {
  int idx = blockIdx.x * blockDim.x + threadIdx.x;
  if (idx >= L_SEQ * TAGS) return;
  int t = idx / TAGS, tag = idx % TAGS;
  const float* __restrict__ w  = W_out + tag * (2 * H2D);
  const float* __restrict__ hf = hs_f + (long long)t * H2D;
  const float* __restrict__ hb = hs_b + (long long)(L_SEQ - 1 - t) * H2D;
  float acc = b_out[tag];
#pragma unroll 4
  for (int u = 0; u < H2D; ++u) acc += hf[u] * w[u];
#pragma unroll 4
  for (int u = 0; u < H2D; ++u) acc += hb[u] * w[H2D + u];
  feats[idx] = acc;
}

// ---------------------------------------------------------------------------
// Phase 4: Viterbi. Single wave; fv + transitions in LDS; bptrs in ws.
// START=1, STOP=0. Serial backtrace on lane 0.
// ---------------------------------------------------------------------------
__global__ void viterbi_kernel(const float* __restrict__ feats,
                               const float* __restrict__ trans,
                               float* __restrict__ out,
                               unsigned char* __restrict__ bptr) {
  __shared__ float fv_s[TAGS];
  __shared__ float trans_s[TAGS * TAGS];
  const int lane = threadIdx.x;
  for (int i = lane; i < TAGS * TAGS; i += 32) trans_s[i] = trans[i];
  if (lane < TAGS) fv_s[lane] = (lane == 1) ? 0.0f : NEGV;  // START=1
  __syncthreads();
  for (int t = 0; t < L_SEQ; ++t) {
    float best = -3.4e38f; int bestp = 0;
    if (lane < TAGS) {
      for (int p = 0; p < TAGS; ++p) {
        float v = fv_s[p] + trans_s[lane * TAGS + p];
        if (v > best) { best = v; bestp = p; }
      }
    }
    __syncthreads();
    if (lane < TAGS) {
      fv_s[lane] = best + feats[t * TAGS + lane];
      bptr[t * TAGS + lane] = (unsigned char)bestp;
    }
    __syncthreads();
  }
  if (lane == 0) {
    float bestv = -3.4e38f; int bt = 0;
    for (int tag = 0; tag < TAGS; ++tag) {          // terminal: + trans[STOP=0]
      float v = fv_s[tag] + trans_s[0 * TAGS + tag];
      if (v > bestv) { bestv = v; bt = tag; }
    }
    out[0] = bestv;                                 // score
    out[1 + (L_SEQ - 1)] = (float)bt;               // path[L-1] = best_last
    int tag = bt;
    for (int t = L_SEQ - 1; t >= 1; --t) {          // path[t-1] = outs[t]
      tag = bptr[t * TAGS + tag];
      out[1 + (t - 1)] = (float)tag;
    }
  }
}

// ---------------------------------------------------------------------------
extern "C" void kernel_launch(void* const* d_in, const int* in_sizes, int n_in,
                              void* d_out, int out_size, void* d_ws, size_t ws_size,
                              hipStream_t stream) {
  (void)in_sizes; (void)n_in; (void)out_size; (void)ws_size;
  const int*   sentence   = (const int*)  d_in[0];
  const float* embed_tab  = (const float*)d_in[1];
  const float* W_ih_f     = (const float*)d_in[2];
  const float* W_hh_f     = (const float*)d_in[3];
  const float* b_ih_f     = (const float*)d_in[4];
  const float* b_hh_f     = (const float*)d_in[5];
  const float* W_ih_b     = (const float*)d_in[6];
  const float* W_hh_b     = (const float*)d_in[7];
  const float* b_ih_b     = (const float*)d_in[8];
  const float* b_hh_b     = (const float*)d_in[9];
  const float* W_out      = (const float*)d_in[10];
  const float* b_out      = (const float*)d_in[11];
  const float* trans      = (const float*)d_in[12];
  const float* h0         = (const float*)d_in[13];
  const float* c0         = (const float*)d_in[14];
  float* out = (float*)d_out;

  char* ws = (char*)d_ws;
  size_t o = 0;
  _Float16* emb_h = (_Float16*)(ws + o); o += (size_t)L_SEQ * EDIM * 2;  // 2 MB
  _Float16* Wf_h  = (_Float16*)(ws + o); o += (size_t)G4 * EDIM * 2;     // 512 KB
  _Float16* Wb_h  = (_Float16*)(ws + o); o += (size_t)G4 * EDIM * 2;     // 512 KB
  float* xp_f     = (float*)(ws + o);    o += (size_t)L_SEQ * G4 * 4;    // 16 MB
  float* xp_b     = (float*)(ws + o);    o += (size_t)L_SEQ * G4 * 4;    // 16 MB
  float* hs_f     = (float*)(ws + o);    o += (size_t)L_SEQ * H2D * 4;   // 4 MB
  float* hs_b     = (float*)(ws + o);    o += (size_t)L_SEQ * H2D * 4;   // 4 MB
  float* feats    = (float*)(ws + o);    o += (size_t)L_SEQ * TAGS * 4;  // 320 KB
  unsigned char* bptr = (unsigned char*)(ws + o);                        // 80 KB

  // 1a) gather + f16 conversion
  {
    int total = L_SEQ * EDIM + 2 * G4 * EDIM;       // 1,572,864
    prep_kernel<<<(total + 255) / 256, 256, 0, stream>>>(
        sentence, embed_tab, W_ih_f, W_ih_b, emb_h, Wf_h, Wb_h);
  }
  // 1b) WMMA input projections: 32768 tile-waves, 8 waves/block
  inproj_kernel<<<4096, 256, 0, stream>>>(emb_h, Wf_h, Wb_h,
                                          b_ih_f, b_hh_f, b_ih_b, b_hh_b,
                                          xp_f, xp_b);
  // 2) sequential scan, one block per direction
  lstm_scan_kernel<<<2, 1024, 0, stream>>>(xp_f, xp_b, W_hh_f, W_hh_b,
                                           h0, c0, hs_f, hs_b);
  // 3) output projection
  outproj_kernel<<<(L_SEQ * TAGS + 255) / 256, 256, 0, stream>>>(
      hs_f, hs_b, W_out, b_out, feats);
  // 4) Viterbi decode
  viterbi_kernel<<<1, 32, 0, stream>>>(feats, trans, out, bptr);
}
